// BasicBlock_6983616824350
// MI455X (gfx1250) — 8-bit matrix operand order repaired and checked by reading, compiled, not run on hardware
//
#include <hip/hip_runtime.h>
#include <stdint.h>

typedef __attribute__((ext_vector_type(8))) int v8i;

#define B_   16
#define C_   256
#define H_   56
#define W_   56
#define HW_  (H_*W_)
#define HP_  58
#define CNT_ (B_*HW_)


__global__ void k_rownorm(const float* __restrict__ Wm, float* __restrict__ nrm,
                          int rows, int cols) {
    int j = blockIdx.x * blockDim.x + threadIdx.x;
    if (j >= rows) return;
    const float* r = Wm + (size_t)j * cols;
    float s = 0.f;
    for (int e = 0; e < cols; ++e) s += r[e] * r[e];
    nrm[j] = sqrtf(s + 1e-6f);
}

__global__ void k_fc2(const float* __restrict__ z, const float* __restrict__ W2,
                      const float* __restrict__ nrm, float* __restrict__ h) {
    int t = blockIdx.x * blockDim.x + threadIdx.x;
    if (t >= 256 * 1024) return;
    int r = t >> 10, j = t & 1023;
    const float* zr = z  + r * 64;
    const float* wj = W2 + j * 64;
    float s = 0.f;
    #pragma unroll 8
    for (int e = 0; e < 64; ++e) s += zr[e] * wj[e];
    h[t] = s / nrm[j];
}

__global__ void k_fc3_pack(const float* __restrict__ h, const float* __restrict__ W3,
                           signed char* __restrict__ Wb) {
    int t = blockIdx.x * blockDim.x + threadIdx.x;
    if (t >= 4096 * 144) return;
    int r2 = t / 144, f = t - r2 * 144;
    int no  = r2 >> 8;
    int rem = r2 & 255;
    int ni  = rem >> 4;
    int i   = rem & 15;
    const float* hr = h  + ((no * 16 + ni) * 1024 + i * 64);
    const float* w3 = W3 + f * 64;
    float s = 0.f;
    #pragma unroll 8
    for (int e = 0; e < 64; ++e) s += hr[e] * w3[e];
    int j = f / 9, q = f - j * 9;
    int co = no * 16 + i, ci = ni * 16 + j;
    Wb[((size_t)q * 256 + co) * 256 + ci] = (s > 0.f) ? 1 : ((s < 0.f) ? -1 : 0);
}

__global__ void k_biasgen(const float* __restrict__ z, const float* __restrict__ W0,
                          float* __restrict__ biasf) {
    int t = blockIdx.x * blockDim.x + threadIdx.x;
    if (t >= 256) return;
    int no = t >> 4, i = t & 15;
    const float* zr = z  + no * 1024;
    const float* w0 = W0 + i  * 1024;
    float s = 0.f;
    for (int k = 0; k < 1024; ++k) s += zr[k] * w0[k];
    biasf[t] = (s > 0.f) ? 1.f : ((s < 0.f) ? -1.f : 0.f);
}


__global__ void k_binarize_x(const float* __restrict__ x, signed char* __restrict__ Xb) {
    int t = blockIdx.x * blockDim.x + threadIdx.x;
    if (t >= B_ * HP_ * HP_ * C_) return;
    int c  = t & 255;
    int p  = t >> 8;
    int ww = p % HP_;  p /= HP_;
    int hh = p % HP_;
    int b  = p / HP_;
    signed char v = 0;
    if (hh >= 1 && hh <= H_ && ww >= 1 && ww <= W_) {
        float f = x[(((size_t)b * C_ + c) * H_ + (hh - 1)) * W_ + (ww - 1)];
        v = (f > 0.f) ? 1 : ((f < 0.f) ? -1 : 0);
    }
    Xb[t] = v;
}

__global__ void k_bn_sign(const float* __restrict__ raw, signed char* __restrict__ Xb,
                          const float* __restrict__ mean, const float* __restrict__ var,
                          const float* __restrict__ gamma, const float* __restrict__ beta) {
    int t = blockIdx.x * blockDim.x + threadIdx.x;
    if (t >= B_ * HP_ * HP_ * C_) return;
    int c  = t & 255;
    int p  = t >> 8;
    int ww = p % HP_;  p /= HP_;
    int hh = p % HP_;
    int b  = p / HP_;
    signed char v = 0;
    if (hh >= 1 && hh <= H_ && ww >= 1 && ww <= W_) {
        float y = raw[((size_t)b * C_ + c) * HW_ + (hh - 1) * W_ + (ww - 1)];
        float u = gamma[c] * (y - mean[c]) * rsqrtf(var[c] + 1e-5f) + beta[c];
        v = (u > 0.f) ? 1 : ((u < 0.f) ? -1 : 0);
    }
    Xb[t] = v;
}

__global__ __launch_bounds__(128)
void k_conv_bin_wmma(const signed char* __restrict__ Xb,
                     const signed char* __restrict__ Wb,
                     const float* __restrict__ biasf,
                     float* __restrict__ outRaw) {
    const int b      = blockIdx.z;
    const int mBlock = blockIdx.y * 64;
    const int nBlock = blockIdx.x * 64;
    const int lane   = threadIdx.x & 31;
    const int wave   = threadIdx.x >> 5;
    const int waveM  = (wave >> 1) * 32;
    const int waveN  = (wave & 1) * 32;
    const int half   = lane >> 4;
    const int l16    = lane & 15;

    v8i acc00 = {}, acc01 = {}, acc10 = {}, acc11 = {};

    size_t pixBase[2];
    #pragma unroll
    for (int sm = 0; sm < 2; ++sm) {
        int p = mBlock + waveM + sm * 16 + l16;
        int hh = p / W_, ww = p - hh * W_;
        pixBase[sm] = (((size_t)b * HP_ + hh) * HP_ + ww) * C_;
    }
    const int n0    = nBlock + waveN + l16;
    const int aKoff = half * 16;
    const int bKoff = half * 16;

    for (int q = 0; q < 9; ++q) {
        const size_t tapOff = (size_t)((q / 3) * HP_ + (q % 3)) * C_;
        const signed char* wb0 = Wb + ((size_t)q * C_ + n0) * C_;
        const signed char* wb1 = wb0 + 16 * C_;
        #pragma unroll
        for (int cib = 0; cib < 4; ++cib) {
            const int cbase = cib * 64;
            v8i a0, a1, b0, b1;
            {
                const signed char* p0 = Xb + pixBase[0] + tapOff + cbase + aKoff;
                const signed char* p1 = Xb + pixBase[1] + tapOff + cbase + aKoff;
                union { uint64_t u[4]; v8i v; } A0, A1;
                #pragma unroll
                for (int P = 0; P < 4; ++P) {
                    A0.u[P] = *(const uint64_t*)(p0 + (P >> 1) * 32 + (P & 1) * 8);
                    A1.u[P] = *(const uint64_t*)(p1 + (P >> 1) * 32 + (P & 1) * 8);
                }
                a0 = A0.v; a1 = A1.v;
            }
            {
                union { uint4 u[2]; v8i v; } Bx, By;
                const signed char* w0 = wb0 + cbase + bKoff;
                const signed char* w1 = wb1 + cbase + bKoff;
                Bx.u[0] = *(const uint4*)(w0);
                Bx.u[1] = *(const uint4*)(w0 + 32);
                By.u[0] = *(const uint4*)(w1);
                By.u[1] = *(const uint4*)(w1 + 32);
                b0 = Bx.v; b1 = By.v;
            }
            acc00 = __builtin_amdgcn_wmma_i32_16x16x64_iu8(true, a0, true, b0, acc00, false, false);
            acc01 = __builtin_amdgcn_wmma_i32_16x16x64_iu8(true, a0, true, b1, acc01, false, false);
            acc10 = __builtin_amdgcn_wmma_i32_16x16x64_iu8(true, a1, true, b0, acc10, false, false);
            acc11 = __builtin_amdgcn_wmma_i32_16x16x64_iu8(true, a1, true, b1, acc11, false, false);
        }
    }

    #pragma unroll
    for (int sm = 0; sm < 2; ++sm) {
        const int pb = mBlock + waveM + sm * 16 + half * 8;
        #pragma unroll
        for (int sn = 0; sn < 2; ++sn) {
            const int n = nBlock + waveN + sn * 16 + l16;
            const float bv = biasf[n];
            const v8i a = (sm == 0) ? (sn == 0 ? acc00 : acc01)
                                    : (sn == 0 ? acc10 : acc11);
            float* op = outRaw + ((size_t)b * C_ + n) * HW_ + pb;
            #pragma unroll
            for (int r = 0; r < 8; ++r)
                op[r] = (float)a[r] + bv;
        }
    }
}


__global__ void k_stats(const float* __restrict__ raw,
                        float* __restrict__ mean, float* __restrict__ var) {
    __shared__ float s1[256];
    __shared__ float s2[256];
    int c = blockIdx.x, t = threadIdx.x;
    float a = 0.f, b = 0.f;
    for (int i = t; i < CNT_; i += 256) {
        int bb = i / HW_, s = i - bb * HW_;
        float v = raw[((size_t)bb * C_ + c) * HW_ + s];
        a += v; b += v * v;
    }
    s1[t] = a; s2[t] = b;
    __syncthreads();
    for (int o = 128; o > 0; o >>= 1) {
        if (t < o) { s1[t] += s1[t + o]; s2[t] += s2[t + o]; }
        __syncthreads();
    }
    if (t == 0) {
        float m = s1[0] / (float)CNT_;
        mean[c] = m;
        var[c]  = s2[0] / (float)CNT_ - m * m;
    }
}


__global__ void k_bn_residual(float* __restrict__ out, const float* __restrict__ x,
                              const float* __restrict__ mean, const float* __restrict__ var,
                              const float* __restrict__ gamma, const float* __restrict__ beta,
                              int n) {
    int i = blockIdx.x * blockDim.x + threadIdx.x;
    if (i >= n) return;
    int c = (i / HW_) & 255;
    float y = out[i];
    out[i] = gamma[c] * (y - mean[c]) * rsqrtf(var[c] + 1e-5f) + beta[c] + x[i];
}


extern "C" void kernel_launch(void* const* d_in, const int* in_sizes, int n_in,
                              void* d_out, int out_size, void* d_ws, size_t ws_size,
                              hipStream_t stream) {
    const float* x      = (const float*)d_in[0];
    const float* z1     = (const float*)d_in[1];
    const float* W0_1   = (const float*)d_in[2];
    const float* W2_1   = (const float*)d_in[3];
    const float* W3_1   = (const float*)d_in[4];
    const float* gamma1 = (const float*)d_in[5];
    const float* beta1  = (const float*)d_in[6];
    const float* z2     = (const float*)d_in[7];
    const float* W0_2   = (const float*)d_in[8];
    const float* W2_2   = (const float*)d_in[9];
    const float* W3_2   = (const float*)d_in[10];
    const float* gamma2 = (const float*)d_in[11];
    const float* beta2  = (const float*)d_in[12];
    float* out = (float*)d_out;

    char* ws = (char*)d_ws;
    size_t off = 0;
    signed char* Xb  = (signed char*)(ws + off); off += (size_t)B_ * HP_ * HP_ * C_;
    float*       h   = (float*)(ws + off);       off += 256 * 1024 * 4;
    float*       nrm = (float*)(ws + off);       off += 1024 * 4;
    signed char* Wb  = (signed char*)(ws + off); off += (size_t)9 * 256 * 256;
    off = (off + 255) & ~(size_t)255;
    float*       bia = (float*)(ws + off);       off += 256 * 4;
    float*       mea = (float*)(ws + off);       off += 256 * 4;
    float*       var = (float*)(ws + off);       off += 256 * 4;

    const int padN = B_ * HP_ * HP_ * C_;
    const int padBlocks = (padN + 255) / 256;
    const int outN = B_ * C_ * HW_;
    dim3 convGrid(4, HW_ / 64, B_), convBlk(128);

    k_rownorm   <<<4,    256, 0, stream>>>(W2_1, nrm, 1024, 64);
    k_fc2       <<<1024, 256, 0, stream>>>(z1, W2_1, nrm, h);
    k_fc3_pack  <<<2304, 256, 0, stream>>>(h, W3_1, Wb);
    k_biasgen   <<<1,    256, 0, stream>>>(z1, W0_1, bia);
    k_binarize_x<<<padBlocks, 256, 0, stream>>>(x, Xb);
    k_conv_bin_wmma<<<convGrid, convBlk, 0, stream>>>(Xb, Wb, bia, out);
    k_stats     <<<256,  256, 0, stream>>>(out, mea, var);
    k_bn_sign   <<<padBlocks, 256, 0, stream>>>(out, Xb, mea, var, gamma1, beta1);

    k_rownorm   <<<4,    256, 0, stream>>>(W2_2, nrm, 1024, 64);
    k_fc2       <<<1024, 256, 0, stream>>>(z2, W2_2, nrm, h);
    k_fc3_pack  <<<2304, 256, 0, stream>>>(h, W3_2, Wb);
    k_biasgen   <<<1,    256, 0, stream>>>(z2, W0_2, bia);
    k_conv_bin_wmma<<<convGrid, convBlk, 0, stream>>>(Xb, Wb, bia, out);
    k_stats     <<<256,  256, 0, stream>>>(out, mea, var);
    k_bn_residual<<<(outN + 255) / 256, 256, 0, stream>>>(out, x, mea, var, gamma2, beta2, outN);
}
